// CANN_18854906429893
// MI455X (gfx1250) — matrix operand order repaired and checked by reading, not compiled, not run on hardware
//
#include <hip/hip_runtime.h>
#include <hip/hip_bf16.h>

#define NODES   50000
#define EDGES   400000
#define DD      128
#define HH      4
#define NLAYER  3
#define DHD     512      // DD*HH
#define ODIM    64
#define NEG_SLOPE 0.01f
#define LN_EPS  1e-5f

typedef __bf16 bf16;
typedef __attribute__((ext_vector_type(16))) __bf16 v16bf;
typedef __attribute__((ext_vector_type(8)))  float  v8f;

// ---------- helpers ----------
__device__ __forceinline__ float wave_sum32(float v) {
#pragma unroll
  for (int m = 16; m >= 1; m >>= 1) v += __shfl_xor(v, m, 32);
  return v;
}
// monotonic float<->int mapping for atomicMax on floats
__device__ __forceinline__ int f2ord(float f) {
  int i = __float_as_int(f);
  return i >= 0 ? i : (i ^ 0x7FFFFFFF);
}
__device__ __forceinline__ float ord2f(int i) {
  return __int_as_float(i >= 0 ? i : (i ^ 0x7FFFFFFF));
}

// A fragment (ISA 16-bit A 16x32 layout): two contiguous 8-float runs per lane.
__device__ __forceinline__ v16bf load_a_frag(const float* __restrict__ Arow, int kb, int half) {
  const float* p0 = Arow + kb * 32 + 8 * half;
  float4 a0 = *(const float4*)(p0);
  float4 a1 = *(const float4*)(p0 + 4);
  float4 a2 = *(const float4*)(p0 + 16);
  float4 a3 = *(const float4*)(p0 + 20);
  v16bf a;
  a[0] = (bf16)a0.x; a[1] = (bf16)a0.y; a[2] = (bf16)a0.z; a[3] = (bf16)a0.w;
  a[4] = (bf16)a1.x; a[5] = (bf16)a1.y; a[6] = (bf16)a1.z; a[7] = (bf16)a1.w;
  a[8] = (bf16)a2.x; a[9] = (bf16)a2.y; a[10] = (bf16)a2.z; a[11] = (bf16)a2.w;
  a[12] = (bf16)a3.x; a[13] = (bf16)a3.y; a[14] = (bf16)a3.z; a[15] = (bf16)a3.w;
  return a;
}

// ---------- pack f32 weights into bf16 WMMA B-fragment order ----------
// Bp layout: [tn][kb][lane][e] so each lane reads 16 contiguous bf16 (32B) per frag.
// Value at (tn,kb,lane,e) = B[k*Nout + col], k = kb*32 + (e&7) + 8*(lane>>4) + 16*(e>>3),
// col = tn*16 + (lane&15).
__global__ void pack_b_kernel(const float* __restrict__ B, bf16* __restrict__ Bp,
                              int KB, int Nout) {
  int total = KB * (Nout >> 4) * 512;
  for (int idx = blockIdx.x * blockDim.x + threadIdx.x; idx < total;
       idx += gridDim.x * blockDim.x) {
    int e    = idx & 15;
    int lane = (idx >> 4) & 31;
    int rem  = idx >> 9;
    int kb   = rem % KB;
    int tn   = rem / KB;
    int half = lane >> 4;
    int k    = kb * 32 + (e & 7) + 8 * half + ((e >> 3) << 4);
    int col  = tn * 16 + (lane & 15);
    Bp[idx] = (bf16)B[(size_t)k * Nout + col];
  }
}

// ---------- WMMA GEMM: each wave computes a 16x64 strip (4 N-tiles) ----------
// A: f32 row-major [M,K], converted to bf16 in-register (f32 accumulate).
// Bp: pre-packed bf16 fragments; K compile-time => fully unrolled, software-
// pipelined one stage ahead so loads for kb+1 overlap the 4 WMMAs of kb.
template <int K, bool HAS_BIAS, bool HAS_RESID, bool RELU>
__global__ __launch_bounds__(256) void wmma_gemm_kernel(
    const float* __restrict__ A, const bf16* __restrict__ Bp,
    const float* __restrict__ bias, const float* __restrict__ resid,
    float* __restrict__ C, int M, int Nout) {
  constexpr int KB = K / 32;
  int wave = (blockIdx.x * blockDim.x + threadIdx.x) >> 5;
  int lane = threadIdx.x & 31;
  int groups_n = Nout >> 6;              // groups of 4 N-tiles
  int tm = wave / groups_n;
  int tg = wave - tm * groups_n;
  if (tm * 16 >= M) return;
  int half = lane >> 4;
  int mr = lane & 15;                    // A row in tile / C col in tile

  v8f acc[4];
#pragma unroll
  for (int t = 0; t < 4; ++t) {
    float bv = 0.0f;
    if constexpr (HAS_BIAS) bv = bias[(tg * 4 + t) * 16 + mr];
#pragma unroll
    for (int j = 0; j < 8; ++j) acc[t][j] = bv;
  }

  const float* Arow = A + (size_t)(tm * 16 + mr) * K;
  const bf16* Bw = Bp + ((size_t)(tg * 4) * KB * 32 + lane) * 16;

  // prologue: fragments for kb = 0
  v16bf aCur = load_a_frag(Arow, 0, half);
  v16bf bCur[4], bNxt[4];
#pragma unroll
  for (int t = 0; t < 4; ++t) bCur[t] = *(const v16bf*)(Bw + ((size_t)t * KB + 0) * 512);

#pragma unroll
  for (int kb = 0; kb < KB; ++kb) {
    v16bf aNxt;
    if (kb + 1 < KB) {                   // issue next-step loads first
      aNxt = load_a_frag(Arow, kb + 1, half);
#pragma unroll
      for (int t = 0; t < 4; ++t)
        bNxt[t] = *(const v16bf*)(Bw + ((size_t)t * KB + (kb + 1)) * 512);
    }
#pragma unroll
    for (int t = 0; t < 4; ++t)
      acc[t] = __builtin_amdgcn_wmma_f32_16x16x32_bf16(false, aCur, false, bCur[t],
                                                       (short)0, acc[t], false, false);
    if (kb + 1 < KB) {
      aCur = aNxt;
#pragma unroll
      for (int t = 0; t < 4; ++t) bCur[t] = bNxt[t];
    }
  }

#pragma unroll
  for (int t = 0; t < 4; ++t) {
    int col = (tg * 4 + t) * 16 + mr;
#pragma unroll
    for (int j = 0; j < 8; ++j) {
      int row = tm * 16 + j + 8 * half;
      float v = acc[t][j];
      if constexpr (HAS_RESID) v += resid[(size_t)row * Nout + col];
      if constexpr (RELU) v = v > 0.0f ? v : 0.0f;
      C[(size_t)row * Nout + col] = v;
    }
  }
}

// ---------- per-layer small precomputation (rank-1 / attention folds) ----------
__global__ void precompute_layer_kernel(const float* __restrict__ Wq_l, const float* __restrict__ att_l,
                                        const float* __restrict__ wkw, const float* __restrict__ wkb,
                                        const float* __restrict__ gw, const float* __restrict__ gb,
                                        float* __restrict__ wq_eff, float* __restrict__ gv0,
                                        float* __restrict__ gv1, float* __restrict__ c0,
                                        float* __restrict__ c1) {
  int tid = threadIdx.x;
  if (tid < HH * DD) {
    int h = tid >> 7, dd = tid & 127;
    float s = 0.f;
    const float* aq = att_l + h * 3 * DD + DD;
    for (int d = 0; d < DD; ++d) s += Wq_l[(size_t)dd * DHD + h * DD + d] * aq[d];
    wq_eff[h * DD + dd] = s;
  } else if (tid < HH * DD + DD) {
    int d = tid - HH * DD;
    float s1 = 0.f, s0 = 0.f;
    for (int dd = 0; dd < DD; ++dd) {
      float g = gw[(size_t)dd * DD + d];
      s1 += wkw[dd] * g;
      s0 += wkb[dd] * g;
    }
    gv1[d] = s1;
    gv0[d] = s0 + gb[d];
  } else if (tid < HH * DD + DD + HH) {
    int h = tid - (HH * DD + DD);
    float s1 = 0.f, s0 = 0.f;
    const float* ak = att_l + h * 3 * DD;
    for (int d = 0; d < DD; ++d) { s1 += wkw[d] * ak[d]; s0 += wkb[d] * ak[d]; }
    c1[h] = s1;
    c0[h] = s0;
  }
}

// ---------- per-node scores + softmax-state init (wave per node) ----------
__global__ void node_score_kernel(const float* __restrict__ h, const float* __restrict__ v,
                                  const float* __restrict__ Kn, const float* __restrict__ att_l,
                                  const float* __restrict__ wq_eff, const float* __restrict__ c0,
                                  const float* __restrict__ c1, float* __restrict__ sdst,
                                  float* __restrict__ ssrc, int* __restrict__ mx,
                                  float* __restrict__ sm) {
  int n = (blockIdx.x * blockDim.x + threadIdx.x) >> 5;
  if (n >= NODES) return;
  int lane = threadIdx.x & 31;
  int d0 = lane * 4;
  const float4 hv = *(const float4*)(h + (size_t)n * DD + d0);
  float kn = Kn[n];
#pragma unroll
  for (int hh = 0; hh < HH; ++hh) {
    const float4 wq = *(const float4*)(wq_eff + hh * DD + d0);
    const float4 vv = *(const float4*)(v + (size_t)n * DHD + hh * DD + d0);
    const float4 av = *(const float4*)(att_l + hh * 3 * DD + 2 * DD + d0);
    float pq = hv.x * wq.x + hv.y * wq.y + hv.z * wq.z + hv.w * wq.w;
    float pv = vv.x * av.x + vv.y * av.y + vv.z * av.z + vv.w * av.w;
    pq = wave_sum32(pq);
    pv = wave_sum32(pv);
    if (lane == 0) {
      sdst[n * HH + hh] = kn * c1[hh] + c0[hh] + pq;
      ssrc[n * HH + hh] = pv;
      mx[n * HH + hh] = f2ord(__int_as_float(0xFF800000));  // ordered(-inf)
      sm[n * HH + hh] = 0.0f;
    }
  }
}

// ---------- edge pass 1: leaky score + segment max ----------
__global__ void edge_score_kernel(const int* __restrict__ src, const int* __restrict__ dst,
                                  const float* __restrict__ sdst, const float* __restrict__ ssrc,
                                  float* __restrict__ ebuf, int* __restrict__ mx) {
  int e = blockIdx.x * blockDim.x + threadIdx.x;
  if (e >= EDGES) return;
  int s = src[e], d = dst[e];
  float4 sd = *(const float4*)(sdst + (size_t)d * HH);
  float4 ss = *(const float4*)(ssrc + (size_t)s * HH);
  float sc[4] = {sd.x + ss.x, sd.y + ss.y, sd.z + ss.z, sd.w + ss.w};
  float4 out;
  float* op = &out.x;
#pragma unroll
  for (int hh = 0; hh < HH; ++hh) {
    float x = sc[hh];
    x = x > 0.f ? x : NEG_SLOPE * x;
    op[hh] = x;
    atomicMax(&mx[d * HH + hh], f2ord(x));
  }
  *(float4*)(ebuf + (size_t)e * HH) = out;
}

// ---------- edge pass 2: exp + segment sum ----------
__global__ void edge_exp_kernel(const int* __restrict__ dst, float* __restrict__ ebuf,
                                const int* __restrict__ mx, float* __restrict__ sm) {
  int e = blockIdx.x * blockDim.x + threadIdx.x;
  if (e >= EDGES) return;
  int d = dst[e];
  float4 ev = *(const float4*)(ebuf + (size_t)e * HH);
  float* ep = &ev.x;
#pragma unroll
  for (int hh = 0; hh < HH; ++hh) {
    float m = ord2f(mx[d * HH + hh]);
    float ex = expf(ep[hh] - m);
    ep[hh] = ex;
    atomicAdd(&sm[d * HH + hh], ex);
  }
  *(float4*)(ebuf + (size_t)e * HH) = ev;
}

// ---------- edge pass 3: alpha * v[src] scatter-add (wave per (edge,head)) ----------
__global__ void edge_agg_kernel(const int* __restrict__ src, const int* __restrict__ dst,
                                const float* __restrict__ ebuf, const float* __restrict__ sm,
                                const float* __restrict__ v, float* __restrict__ agg) {
  long long wid = ((long long)blockIdx.x * blockDim.x + threadIdx.x) >> 5;
  if (wid >= (long long)EDGES * HH) return;
  int e = (int)(wid >> 2);
  int hh = (int)(wid & 3);
  int lane = threadIdx.x & 31;
  int s = src[e], d = dst[e];
  float alpha = ebuf[(size_t)e * HH + hh] / sm[d * HH + hh];
  int d0 = hh * DD + lane * 4;
  float4 vv = *(const float4*)(v + (size_t)s * DHD + d0);
  float* dp = agg + (size_t)d * DHD + d0;
  atomicAdd(dp + 0, alpha * vv.x);
  atomicAdd(dp + 1, alpha * vv.y);
  atomicAdd(dp + 2, alpha * vv.z);
  atomicAdd(dp + 3, alpha * vv.w);
}

// ---------- layernorm + rank-1 gate + relu (wave per node), writes h in place ----------
__global__ void ln_gate_kernel(const float* __restrict__ out, const float* __restrict__ Kn,
                               const float* __restrict__ lng, const float* __restrict__ lnb,
                               const float* __restrict__ gv0, const float* __restrict__ gv1,
                               const float* __restrict__ bl, float* __restrict__ hout) {
  int n = (blockIdx.x * blockDim.x + threadIdx.x) >> 5;
  if (n >= NODES) return;
  int lane = threadIdx.x & 31;
  int d0 = lane * 4;
  float4 o4 = *(const float4*)(out + (size_t)n * DD + d0);
  float o[4] = {o4.x, o4.y, o4.z, o4.w};
  float s = wave_sum32(o[0] + o[1] + o[2] + o[3]);
  float mu = s * (1.0f / DD);
  float sq = 0.f;
#pragma unroll
  for (int c = 0; c < 4; ++c) { float t = o[c] - mu; sq += t * t; }
  sq = wave_sum32(sq);
  float inv = rsqrtf(sq * (1.0f / DD) + LN_EPS);
  float kn = Kn[n];
#pragma unroll
  for (int c = 0; c < 4; ++c) {
    int d = d0 + c;
    float norm = (o[c] - mu) * inv * lng[d] + lnb[d];
    float g = 1.0f / (1.0f + expf(-(kn * gv1[d] + gv0[d])));
    float val = norm * (1.0f + g) + bl[d];
    hout[(size_t)n * DD + d] = val > 0.f ? val : 0.f;
  }
}

// ---------- host ----------
extern "C" void kernel_launch(void* const* d_in, const int* in_sizes, int n_in,
                              void* d_out, int out_size, void* d_ws, size_t ws_size,
                              hipStream_t stream) {
  const float* x      = (const float*)d_in[0];
  const int*   ei     = (const int*)d_in[1];
  const float* Kn     = (const float*)d_in[2];
  const float* W_in   = (const float*)d_in[3];
  const float* b_in   = (const float*)d_in[4];
  const float* Wk_w   = (const float*)d_in[5];
  const float* Wk_b   = (const float*)d_in[6];
  const float* Wq     = (const float*)d_in[7];
  const float* Wv     = (const float*)d_in[8];
  const float* att    = (const float*)d_in[9];
  const float* Wo     = (const float*)d_in[10];
  const float* b_l    = (const float*)d_in[11];
  const float* ln_g   = (const float*)d_in[12];
  const float* ln_b   = (const float*)d_in[13];
  const float* gate_w = (const float*)d_in[14];
  const float* gate_b = (const float*)d_in[15];
  const float* W_out  = (const float*)d_in[16];
  const float* b_out  = (const float*)d_in[17];
  float* y = (float*)d_out;

  const int* src = ei;
  const int* dst = ei + EDGES;

  // workspace carve-up
  char* base = (char*)d_ws;
  size_t off = 0;
  auto alloc = [&](size_t bytes) { size_t o = off; off = (off + bytes + 255) & ~(size_t)255; return (void*)(base + o); };
  float* h_buf  = (float*)alloc((size_t)NODES * DD * 4);
  float* tmp    = (float*)alloc((size_t)NODES * DD * 4);
  float* v_buf  = (float*)alloc((size_t)NODES * DHD * 4);
  float* agg    = (float*)alloc((size_t)NODES * DHD * 4);
  float* sdst   = (float*)alloc((size_t)NODES * HH * 4);
  float* ssrc   = (float*)alloc((size_t)NODES * HH * 4);
  int*   mx     = (int*)  alloc((size_t)NODES * HH * 4);
  float* sm     = (float*)alloc((size_t)NODES * HH * 4);
  float* ebuf   = (float*)alloc((size_t)EDGES * HH * 4);
  bf16* win_bf  = (bf16*)alloc((size_t)DD * DD * 2);
  bf16* wv_bf   = (bf16*)alloc((size_t)NLAYER * DD * DHD * 2);
  bf16* wo_bf   = (bf16*)alloc((size_t)NLAYER * DHD * DD * 2);
  bf16* wout_bf = (bf16*)alloc((size_t)DD * ODIM * 2);
  float* wq_eff = (float*)alloc((size_t)HH * DD * 4);
  float* gv0    = (float*)alloc((size_t)DD * 4);
  float* gv1    = (float*)alloc((size_t)DD * 4);
  float* c0v    = (float*)alloc((size_t)HH * 4);
  float* c1v    = (float*)alloc((size_t)HH * 4);

  auto pack = [&](const float* s, bf16* d, int K, int Nout) {
    int total = (K >> 5) * (Nout >> 4) * 512;
    int blk = 256, grd = (total + blk - 1) / blk;
    hipLaunchKernelGGL(pack_b_kernel, dim3(grd), dim3(blk), 0, stream, s, d, K >> 5, Nout);
  };
  pack(W_in, win_bf, DD, DD);
  for (int l = 0; l < NLAYER; ++l) {
    pack(Wv + (size_t)l * DD * DHD, wv_bf + (size_t)l * DD * DHD, DD, DHD);
    pack(Wo + (size_t)l * DHD * DD, wo_bf + (size_t)l * DHD * DD, DHD, DD);
  }
  pack(W_out, wout_bf, DD, ODIM);

  auto grid_for = [&](int M, int Nout) {
    long long waves = (long long)(M / 16) * (Nout / 64);
    long long threads = waves * 32;
    return (int)((threads + 255) / 256);
  };

  // input projection: h = relu(x @ W_in + b_in)
  hipLaunchKernelGGL((wmma_gemm_kernel<DD, true, false, true>),
                     dim3(grid_for(NODES, DD)), dim3(256), 0, stream,
                     x, win_bf, b_in, (const float*)nullptr, h_buf, NODES, DD);

  const int node_waves_blk = 256;  // 8 waves/block
  int node_grid = (NODES + 7) / 8;
  int edge_grid = (EDGES + 255) / 256;
  long long agg_threads = (long long)EDGES * HH * 32;
  int agg_grid = (int)((agg_threads + 255) / 256);

  for (int l = 0; l < NLAYER; ++l) {
    const float* att_l = att + (size_t)l * HH * 3 * DD;
    hipLaunchKernelGGL(precompute_layer_kernel, dim3(1), dim3(HH * DD + DD + 32), 0, stream,
                       Wq + (size_t)l * DD * DHD, att_l, Wk_w + (size_t)l * DD,
                       Wk_b + (size_t)l * DD, gate_w + (size_t)l * DD * DD,
                       gate_b + (size_t)l * DD, wq_eff, gv0, gv1, c0v, c1v);

    // v = h @ Wv[l]
    hipLaunchKernelGGL((wmma_gemm_kernel<DD, false, false, false>),
                       dim3(grid_for(NODES, DHD)), dim3(256), 0, stream,
                       h_buf, wv_bf + (size_t)l * DD * DHD, (const float*)nullptr,
                       (const float*)nullptr, v_buf, NODES, DHD);

    hipLaunchKernelGGL(node_score_kernel, dim3(node_grid), dim3(node_waves_blk), 0, stream,
                       h_buf, v_buf, Kn, att_l, wq_eff, c0v, c1v, sdst, ssrc, mx, sm);

    hipMemsetAsync(agg, 0, (size_t)NODES * DHD * 4, stream);

    hipLaunchKernelGGL(edge_score_kernel, dim3(edge_grid), dim3(256), 0, stream,
                       src, dst, sdst, ssrc, ebuf, mx);
    hipLaunchKernelGGL(edge_exp_kernel, dim3(edge_grid), dim3(256), 0, stream,
                       dst, ebuf, mx, sm);
    hipLaunchKernelGGL(edge_agg_kernel, dim3(agg_grid), dim3(256), 0, stream,
                       src, dst, ebuf, sm, v_buf, agg);

    // out = agg @ Wo[l] + h (residual)
    hipLaunchKernelGGL((wmma_gemm_kernel<DHD, false, true, false>),
                       dim3(grid_for(NODES, DD)), dim3(256), 0, stream,
                       agg, wo_bf + (size_t)l * DHD * DD, (const float*)nullptr,
                       h_buf, tmp, NODES, DD);

    hipLaunchKernelGGL(ln_gate_kernel, dim3(node_grid), dim3(node_waves_blk), 0, stream,
                       tmp, Kn, ln_g + (size_t)l * DD, ln_b + (size_t)l * DD,
                       gv0, gv1, b_l + (size_t)l * DD, h_buf);
  }

  // y = h @ W_out + b_out
  hipLaunchKernelGGL((wmma_gemm_kernel<DD, true, false, false>),
                     dim3(grid_for(NODES, ODIM)), dim3(256), 0, stream,
                     h_buf, wout_bf, b_out, (const float*)nullptr, y, NODES, ODIM);
}
